// LogMatVecPackedLinear_46273977647109
// MI455X (gfx1250) — 8-bit matrix operand order repaired and checked by reading, compiled, not run on hardware
//
#include <hip/hip_runtime.h>


typedef int   v16i __attribute__((ext_vector_type(16)));
typedef float v8f  __attribute__((ext_vector_type(8)));

#define IN_F   8192
#define OUT_F  8192
#define NROWS  4096
#define KT     64
#define TILE_SHIFT 11
#define BUF_TILES 24
#define BUF_BYTES (BUF_TILES * 2048u)

__device__ __forceinline__ unsigned e4m3_mag(unsigned m) {
  if (m == 0u) return 0u;
  unsigned e = 31u - __clz((int)m);
  return ((e + 7u) << 3) | ((m << (3u - e)) & 7u);
}

__global__ void quant_split_kernel(const float* __restrict__ x,
                                   const float* __restrict__ scale_p,
                                   unsigned char* __restrict__ P0,
                                   unsigned char* __restrict__ P1) {
  unsigned t    = blockIdx.x * blockDim.x + threadIdx.x;
  unsigned v    = t & 15u;
  unsigned L    = (t >> 4) & 31u;
  unsigned tile = t >> 9;
  unsigned kt   = tile & (KT - 1u);
  unsigned rt   = tile >> 6;
  unsigned row  = rt * 16u + (L & 15u);
  unsigned Ks   = kt * 128u + ((v >> 3) << 6) + (((v & 7u) >> 1) << 4)
                + ((v & 1u) << 2) + ((L >> 4) << 3);

  const float inv = 1.0f / scale_p[0];
  const float4 xv = *(const float4*)(x + (size_t)row * IN_F + Ks);
  float xs[4] = {xv.x, xv.y, xv.z, xv.w};

  unsigned d0 = 0u, d1 = 0u;
#pragma unroll
  for (int b = 0; b < 4; ++b) {
    float r = rintf(xs[b] * inv);
    r = fminf(fmaxf(r, -128.0f), 127.0f);
    int q = (int)r;
    int h = q >> 4;
    unsigned l = (unsigned)(q & 15);
    unsigned lb = e4m3_mag(l);
    unsigned hm = (unsigned)(h < 0 ? -h : h);
    unsigned hb = e4m3_mag(hm);
    if (hb) hb += 0x20u;
    if (h < 0) hb |= 0x80u;
    d0 |= hb << (8 * b);
    d1 |= lb << (8 * b);
  }
  size_t off = ((size_t)tile << TILE_SHIFT) + L * 64u + v * 4u;
  *(unsigned*)(P0 + off) = d0;
  *(unsigned*)(P1 + off) = d1;
}

__global__ void weight_decode_kernel(const int* __restrict__ packed,
                                     const int* __restrict__ signs,
                                     unsigned char* __restrict__ W) {
  unsigned t    = blockIdx.x * blockDim.x + threadIdx.x;
  unsigned v    = t & 15u;
  unsigned L    = (t >> 4) & 31u;
  unsigned tile = t >> 9;
  unsigned kt   = tile & (KT - 1u);
  unsigned ct   = tile >> 6;
  unsigned N    = ct * 16u + (L & 15u);
  unsigned K    = kt * 128u + ((v >> 3) << 6) + (((v & 7u) >> 1) << 4) + ((v & 1u) << 2) + ((L >> 4) << 3);

  const int2 pe = *(const int2*)(packed + (size_t)N * (IN_F / 2) + (K >> 1));
  const int4 sg = *(const int4*)(signs + (size_t)N * IN_F + K);
  int pearr[2] = {pe.x, pe.y};
  int sarr[4]  = {sg.x, sg.y, sg.z, sg.w};

  unsigned d = 0u;
#pragma unroll
  for (int b = 0; b < 4; ++b) {
    unsigned pbyte = (unsigned)pearr[b >> 1];
    unsigned nib   = (b & 1) ? ((pbyte >> 4) & 15u) : (pbyte & 15u);
    unsigned wb    = nib ? (nib << 3) : 0x04u;
    if (sarr[b] < 0) wb |= 0x80u;
    d |= wb << (8 * b);
  }
  *(unsigned*)(W + (((size_t)tile << TILE_SHIFT) + L * 64u + v * 4u)) = d;
}

__global__ void __launch_bounds__(256)
log_gemm_wmma_kernel(const unsigned char* __restrict__ P0,
                     const unsigned char* __restrict__ P1,
                     const unsigned char* __restrict__ W,
                     const float* __restrict__ bias,
                     const float* __restrict__ scale_p,
                     const int* __restrict__ minexp_p,
                     float* __restrict__ out) {
  __shared__ unsigned char lds[2u * BUF_BYTES];

  const unsigned t  = threadIdx.x;
  const unsigned L  = t & 31u;
  const unsigned wv = t >> 5;
  const unsigned wr = wv >> 2, wc = wv & 3u;
  const unsigned br = blockIdx.x & 31u;
  const unsigned bc = blockIdx.x >> 5;
  const unsigned rt_base = br * 8u;
  const unsigned ct_base = bc * 8u;
  const unsigned laneoff = L * 64u;

  unsigned long long gaddr[12];
  unsigned           ldsoff[12];
  {
    const unsigned tbyte = (t & 127u) * 16u;
#pragma unroll
    for (int i = 0; i < 12; ++i) {
      unsigned tileIdx = (t >> 7) + 2u * (unsigned)i;
      const unsigned char* base;
      unsigned gt;
      if (tileIdx < 8u)       { base = P0; gt = (rt_base + tileIdx) * KT; }
      else if (tileIdx < 16u) { base = P1; gt = (rt_base + tileIdx - 8u) * KT; }
      else                    { base = W;  gt = (ct_base + tileIdx - 16u) * KT; }
      gaddr[i]  = (unsigned long long)(base + (((size_t)gt) << TILE_SHIFT) + tbyte);
      ldsoff[i] = tileIdx * 2048u + tbyte;
    }
  }
  const unsigned ldsbase = (unsigned)(size_t)(&lds[0]);

  auto issue_copies = [&](unsigned buf) {
    const unsigned dst = ldsbase + buf * BUF_BYTES;
#pragma unroll
    for (int i = 0; i < 12; ++i) {
      asm volatile("global_load_async_to_lds_b128 %0, %1, off"
                   :: "v"(dst + ldsoff[i]), "v"(gaddr[i]) : "memory");
      gaddr[i] += 2048ull;
    }
  };

  v8f acc[4][2];
#pragma unroll
  for (int r = 0; r < 4; ++r)
#pragma unroll
    for (int c = 0; c < 2; ++c) acc[r][c] = (v8f)0.0f;

  issue_copies(0u);

  for (unsigned kt = 0; kt < KT; ++kt) {
    const unsigned cur = kt & 1u;
    asm volatile("s_wait_asynccnt 0x0" ::: "memory");
    __syncthreads();
    if (kt + 1u < KT) issue_copies(1u - cur);

    const unsigned char* buf = &lds[cur * BUF_BYTES];
    const v16i b0 = *(const v16i*)(buf + (16u + wc * 2u + 0u) * 2048u + laneoff);
    const v16i b1 = *(const v16i*)(buf + (16u + wc * 2u + 1u) * 2048u + laneoff);
#pragma unroll
    for (int r = 0; r < 4; ++r) {
      const unsigned j = wr * 4u + (unsigned)r;
      const v16i a0 = *(const v16i*)(buf + j * 2048u + laneoff);
      const v16i a1 = *(const v16i*)(buf + (8u + j) * 2048u + laneoff);
      acc[r][0] = __builtin_amdgcn_wmma_f32_16x16x128_fp8_fp8(a0, b0, (short)0, acc[r][0], false, false);
      acc[r][0] = __builtin_amdgcn_wmma_f32_16x16x128_fp8_fp8(a1, b0, (short)0, acc[r][0], false, false);
      acc[r][1] = __builtin_amdgcn_wmma_f32_16x16x128_fp8_fp8(a0, b1, (short)0, acc[r][1], false, false);
      acc[r][1] = __builtin_amdgcn_wmma_f32_16x16x128_fp8_fp8(a1, b1, (short)0, acc[r][1], false, false);
    }
  }

  const float fs = scale_p[0] * exp2f((float)(minexp_p[0] + 7));
  const unsigned nloc = L & 15u;
  const unsigned mhi  = (L >> 4) * 8u;
#pragma unroll
  for (int c = 0; c < 2; ++c) {
    const unsigned col = (ct_base + wc * 2u + (unsigned)c) * 16u + nloc;
    const float bv = bias[col];
#pragma unroll
    for (int r = 0; r < 4; ++r) {
      const unsigned rowbase = (rt_base + wr * 4u + (unsigned)r) * 16u + mhi;
#pragma unroll
      for (int i = 0; i < 8; ++i) {
        out[(size_t)(rowbase + (unsigned)i) * OUT_F + col] = acc[r][c][i] * fs + bv;
      }
    }
  }
}

extern "C" void kernel_launch(void* const* d_in, const int* in_sizes, int n_in,
                              void* d_out, int out_size, void* d_ws, size_t ws_size,
                              hipStream_t stream) {
  const float* x      = (const float*)d_in[0];
  const int*   packed = (const int*)d_in[1];
  const int*   signs  = (const int*)d_in[2];
  const float* bias   = (const float*)d_in[3];
  const float* scale  = (const float*)d_in[4];
  const int*   minexp = (const int*)d_in[5];
  float* out = (float*)d_out;

  unsigned char* ws = (unsigned char*)d_ws;
  unsigned char* P0 = ws;
  unsigned char* P1 = ws + (32u << 20);
  unsigned char* Wd = ws + (64u << 20);

  quant_split_kernel  <<<32768, 256, 0, stream>>>(x, scale, P0, P1);
  weight_decode_kernel<<<65536, 256, 0, stream>>>(packed, signs, Wd);
  log_gemm_wmma_kernel<<<2048, 256, 0, stream>>>(P0, P1, Wd, bias, scale, minexp, out);
}
